// GroupedSwiGLU_34007551050240
// MI455X (gfx1250) — matrix operand order repaired and checked by reading, not compiled, not run on hardware
//
#include <hip/hip_runtime.h>
#include <hip/hip_bf16.h>

// ---------------------------------------------------------------------------
// Grouped SwiGLU MoE MLP for gfx1250 (MI455X), wave32 + WMMA bf16 + TDM.
//   gate = ragged_dot(X, Wg); up = ragged_dot(X, Wu)
//   h    = silu(gate) * up            (stored bf16 in workspace)
//   out  = probs * ragged_dot(h, Wd)
// Double-buffered LDS pipeline: global loads for tile k+1 are issued before
// the WMMA chain of tile k. In GEMM2 the A (bf16 h) tile is staged by the
// Tensor Data Mover (tensor_load_to_lds + s_wait_tensorcnt), using TDM's
// pad feature to reproduce the bank-skew row pitch in hardware.
// ---------------------------------------------------------------------------

#define E     8
#define HID   2048
#define INTER 1408
#define TOT   8192

#define TM  128      // block tile M
#define TN  64       // block tile N
#define TK  64       // block tile K
#define LDK (TK + 8) // LDS row pitch in bf16 (144B: 16B-aligned, bank-skewed)

typedef __bf16 bf16_t;
typedef __attribute__((ext_vector_type(16))) __bf16 v16bf;
typedef __attribute__((ext_vector_type(8)))  __bf16 v8bf;
typedef __attribute__((ext_vector_type(4)))  __bf16 v4bf;
typedef __attribute__((ext_vector_type(8)))  float  v8f;

typedef __attribute__((ext_vector_type(4))) unsigned int u32x4;
typedef __attribute__((ext_vector_type(4))) int          i32x4;
typedef __attribute__((ext_vector_type(8))) int          i32x8;

union FragAB { v16bf v; v8bf h[2]; };

#if defined(__has_builtin)
#  if __has_builtin(__builtin_amdgcn_tensor_load_to_lds)
#    define HAVE_TDM 1
#  endif
#endif
#ifndef HAVE_TDM
#  define HAVE_TDM 0
#endif
#if __has_include(<hip/amd_detail/amd_gfx1250_TDM.h>)
#  define TDM_ARGS6 1   // amdgpu-toolchain (clang-23): 6-arg builtin
#else
#  define TDM_ARGS6 0   // ROCm 7.2 (clang-22): 5-arg builtin
#endif

__device__ inline void wait_tensorcnt0() {
#if defined(__has_builtin) && __has_builtin(__builtin_amdgcn_s_wait_tensorcnt)
  __builtin_amdgcn_s_wait_tensorcnt(0);
#else
  asm volatile("s_wait_tensorcnt 0x0" ::: "memory");
#endif
}

// ---------------------------------------------------------------------------
// Kernel 0: exclusive prefix sum of tokens_per_expert -> offs[0..E]
// ---------------------------------------------------------------------------
__global__ void moe_offsets_kernel(const int* __restrict__ tpe,
                                   int* __restrict__ offs) {
  if (threadIdx.x == 0) {
    int acc = 0;
    offs[0] = 0;
    for (int i = 0; i < E; ++i) { acc += tpe[i]; offs[i + 1] = acc; }
  }
}

// Map padded tile index bx -> (expert e, first global row row0). e<0 if bx is
// beyond the real (ragged) tile count.
__device__ inline int map_tile(const int* __restrict__ offs, int bx, int* row0) {
  int base = 0;
  for (int i = 0; i < E; ++i) {
    int s = offs[i], t = offs[i + 1];
    int nt = (t - s + TM - 1) / TM;
    if (bx < base + nt) { *row0 = s + (bx - base) * TM; return i; }
    base += nt;
  }
  return -1;
}

// ---------------------------------------------------------------------------
// Kernel 1: fused gate/up grouped GEMM + SiLU-mul, h (bf16) to workspace.
// ---------------------------------------------------------------------------
__global__ __launch_bounds__(256)
void moe_gemm1_kernel(const float* __restrict__ X,
                      const float* __restrict__ Wg,
                      const float* __restrict__ Wu,
                      const int*   __restrict__ offs,
                      bf16_t*      __restrict__ H) {
  __shared__ bf16_t sA [2][TM][LDK];  // X tile, [m][k], bf16
  __shared__ bf16_t sBg[2][TN][LDK];  // Wg tile, TRANSPOSED [n][k]
  __shared__ bf16_t sBu[2][TN][LDK];  // Wu tile, TRANSPOSED [n][k]

  int row0;
  const int e = map_tile(offs, blockIdx.x, &row0);
  if (e < 0) return;
  const int rows = min(offs[e + 1] - row0, TM);
  const int n0   = blockIdx.y * TN;

  const int tid  = threadIdx.x;
  const int lane = tid & 31;
  const int wave = tid >> 5;
  const int wm   = wave & 3;   // 4 waves over M (32 rows each)
  const int wn   = wave >> 2;  // 2 waves over N (32 cols each)
  const int lrow = lane & 15;
  const int half = lane >> 4;

  const float* wg = Wg + (size_t)e * HID * INTER;
  const float* wu = Wu + (size_t)e * HID * INTER;

  // staging registers (tile in flight)
  float4 ra[8];            // A: 8-float chunks (2 float4 each)
  float4 rbg[4], rbu[4];   // B: 4k x 4n sub-block per matrix
  const int arow = tid >> 3, ac8 = (tid & 7) << 3;
  const int bkg  = (tid >> 4) << 2, bng = (tid & 15) << 2;

  auto loadA = [&](int k0) {
#pragma unroll
    for (int j = 0; j < 4; ++j) {
      const int r = arow + j * 32;
      float4 z = make_float4(0.f, 0.f, 0.f, 0.f);
      ra[2 * j] = z; ra[2 * j + 1] = z;
      if (r < rows) {
        const float* p = X + (size_t)(row0 + r) * HID + k0 + ac8;
        ra[2 * j]     = *(const float4*)p;
        ra[2 * j + 1] = *(const float4*)(p + 4);
      }
    }
  };
  auto storeA = [&](int b) {
#pragma unroll
    for (int j = 0; j < 4; ++j) {
      v8bf v;
      v[0] = (bf16_t)ra[2*j].x;   v[1] = (bf16_t)ra[2*j].y;
      v[2] = (bf16_t)ra[2*j].z;   v[3] = (bf16_t)ra[2*j].w;
      v[4] = (bf16_t)ra[2*j+1].x; v[5] = (bf16_t)ra[2*j+1].y;
      v[6] = (bf16_t)ra[2*j+1].z; v[7] = (bf16_t)ra[2*j+1].w;
      *(v8bf*)&sA[b][arow + j * 32][ac8] = v;  // ds_store_b128
    }
  };
  auto loadB = [&](int k0) {
#pragma unroll
    for (int i = 0; i < 4; ++i) {
      const size_t gb = (size_t)(k0 + bkg + i) * INTER + n0 + bng;
      rbg[i] = *(const float4*)(wg + gb);
      rbu[i] = *(const float4*)(wu + gb);
    }
  };
  auto storeB = [&](int b) {
#pragma unroll
    for (int j = 0; j < 4; ++j) {  // register transpose: column j over 4 k-rows
      v4bf g, u;
      g.x = (bf16_t)((const float*)&rbg[0])[j];
      g.y = (bf16_t)((const float*)&rbg[1])[j];
      g.z = (bf16_t)((const float*)&rbg[2])[j];
      g.w = (bf16_t)((const float*)&rbg[3])[j];
      u.x = (bf16_t)((const float*)&rbu[0])[j];
      u.y = (bf16_t)((const float*)&rbu[1])[j];
      u.z = (bf16_t)((const float*)&rbu[2])[j];
      u.w = (bf16_t)((const float*)&rbu[3])[j];
      *(v4bf*)&sBg[b][bng + j][bkg] = g;  // ds_store_b64
      *(v4bf*)&sBu[b][bng + j][bkg] = u;
    }
  };

  const v8f vzero = {};
  v8f accg[2][2], accu[2][2];
#pragma unroll
  for (int i = 0; i < 2; ++i)
#pragma unroll
    for (int j = 0; j < 2; ++j) { accg[i][j] = vzero; accu[i][j] = vzero; }

  auto compute = [&](int b) {
#pragma unroll
    for (int ks = 0; ks < TK; ks += 32) {
      FragAB a[2], bg[2], bu[2];
#pragma unroll
      for (int mt = 0; mt < 2; ++mt) {
        // A 16x32 frag: k = 16*half + (2v+h), contiguous
        const bf16_t* p = &sA[b][wm * 32 + mt * 16 + lrow][ks + half * 16];
        a[mt].h[0] = *(const v8bf*)p;
        a[mt].h[1] = *(const v8bf*)(p + 8);
      }
#pragma unroll
      for (int nt = 0; nt < 2; ++nt) {
        // B 32x16 frag from [n][k] LDS: k = 16*half + (2v+h), contiguous
        const bf16_t* pg = &sBg[b][wn * 32 + nt * 16 + lrow][ks + half * 16];
        bg[nt].h[0] = *(const v8bf*)pg;
        bg[nt].h[1] = *(const v8bf*)(pg + 8);
        const bf16_t* pu = &sBu[b][wn * 32 + nt * 16 + lrow][ks + half * 16];
        bu[nt].h[0] = *(const v8bf*)pu;
        bu[nt].h[1] = *(const v8bf*)(pu + 8);
      }
#pragma unroll
      for (int mt = 0; mt < 2; ++mt)
#pragma unroll
        for (int nt = 0; nt < 2; ++nt) {
          accg[mt][nt] = __builtin_amdgcn_wmma_f32_16x16x32_bf16(
              false, a[mt].v, false, bg[nt].v, (short)0, accg[mt][nt], false, false);
          accu[mt][nt] = __builtin_amdgcn_wmma_f32_16x16x32_bf16(
              false, a[mt].v, false, bu[nt].v, (short)0, accu[mt][nt], false, false);
        }
    }
  };

  // ---- software pipeline over K ------------------------------------------
  const int NK = HID / TK;
  loadA(0); loadB(0);
  storeA(0); storeB(0);
  for (int kt = 0; kt < NK; ++kt) {
    __syncthreads();
    const int cur = kt & 1;
    if (kt + 1 < NK) {
      loadA((kt + 1) * TK);           // global loads issued before the math
      loadB((kt + 1) * TK);
      if (kt + 2 < NK) {              // L2 hints two tiles ahead
        __builtin_prefetch(X + (size_t)(row0 + arow) * HID + (kt + 2) * TK + ac8, 0, 1);
        __builtin_prefetch(wg + (size_t)((kt + 2) * TK + bkg) * INTER + n0 + bng, 0, 1);
        __builtin_prefetch(wu + (size_t)((kt + 2) * TK + bkg) * INTER + n0 + bng, 0, 1);
      }
    }
    compute(cur);                     // WMMA chain hides the load latency
    if (kt + 1 < NK) { storeA(cur ^ 1); storeB(cur ^ 1); }
  }

  // ---- epilogue: h = silu(gate) * up, store bf16 --------------------------
  // C layout: VGPR r -> M = 8*half + r, N = lane%16.
#pragma unroll
  for (int mt = 0; mt < 2; ++mt)
#pragma unroll
    for (int nt = 0; nt < 2; ++nt)
#pragma unroll
      for (int r = 0; r < 8; ++r) {
        const int lm = wm * 32 + mt * 16 + half * 8 + r;
        if (lm < rows) {
          const int   gn = n0 + wn * 32 + nt * 16 + lrow;
          const float g  = accg[mt][nt][r];
          const float u  = accu[mt][nt][r];
          const float hv = (g / (1.0f + __expf(-g))) * u;
          H[(size_t)(row0 + lm) * INTER + gn] = (bf16_t)hv;
        }
      }
}

// ---------------------------------------------------------------------------
// Kernel 2: grouped down-projection, out = probs * (h @ Wd), f32 out.
// A (bf16 h) tile staged by the Tensor Data Mover when available.
// ---------------------------------------------------------------------------
__global__ __launch_bounds__(256)
void moe_gemm2_kernel(const bf16_t* __restrict__ H,
                      const float*  __restrict__ Wd,
                      const float*  __restrict__ probs,
                      const int*    __restrict__ offs,
                      float*        __restrict__ Out) {
  __shared__ bf16_t sA[2][TM][LDK];  // h tile, [m][k] (already bf16)
  __shared__ bf16_t sB[2][TN][LDK];  // Wd tile, TRANSPOSED [n][k]

  int row0;
  const int e = map_tile(offs, blockIdx.x, &row0);
  if (e < 0) return;
  const int rows = min(offs[e + 1] - row0, TM);
  const int n0   = blockIdx.y * TN;

  const int tid  = threadIdx.x;
  const int lane = tid & 31;
  const int wave = tid >> 5;
  const int wm   = wave & 3;
  const int wn   = wave >> 2;
  const int lrow = lane & 15;
  const int half = lane >> 4;

  const float* wd = Wd + (size_t)e * INTER * HID;

  float4 rb[4];            // B staging: 4k x 4n f32 sub-block
  const int bkg = (tid >> 4) << 2, bng = (tid & 15) << 2;

#if HAVE_TDM
  // --- TDM descriptor issue: copy a TMxTK bf16 tile into sA[b] with the
  // bank-skew pad (32 DWORDs data + 4 DWORDs pad per row) done by hardware.
  auto tdmA = [&](int k0, int b) {
    if (wave == 0) {
      const unsigned lds = (unsigned)(uintptr_t)&sA[b][0][0];
      const unsigned long long ga =
          (unsigned long long)(uintptr_t)(H + (size_t)row0 * INTER + k0);
      u32x4 g0 = {};
      g0[0] = 1u;                                   // count=1 (valid user D#)
      g0[1] = lds;                                  // lds_addr
      g0[2] = (unsigned)(ga & 0xFFFFFFFFu);         // global_addr[31:0]
      g0[3] = (unsigned)((ga >> 32) & 0x1FFFFFFu)   // global_addr[56:32]
            | 0x80000000u;                          // type=2 ("image")
      i32x8 g1 = {};
      g1[0] = (1 << 16)        // data_size = 2 bytes (bf16)
            | (1 << 20)        // pad_enable
            | (4 << 22)        // pad_interval: 32 DWORDs (=128B row)
            | (3 << 25);       // pad_amount: 4 DWORDs (=8 bf16 skew)
      g1[1] = (TK << 16);                            // tensor_dim0 = 64
      g1[2] = ((unsigned)rows << 16);                // tensor_dim1 = rows (ragged OOB->0)
      g1[3] = (TK << 16);                            // tile_dim0 = 64
      g1[4] = TM;                                    // tile_dim1 = 128
      g1[5] = INTER;                                 // tensor_dim0_stride
      const i32x4 gz = {};
#if TDM_ARGS6
      const i32x8 z8 = {};
      __builtin_amdgcn_tensor_load_to_lds(g0, g1, gz, gz, z8, 0);
#else
      __builtin_amdgcn_tensor_load_to_lds(g0, g1, gz, gz, 0);
#endif
    }
  };
#else
  v8bf rha[4];
  const int arow = tid >> 3, ac8 = (tid & 7) << 3;
  auto loadA = [&](int k0) {
#pragma unroll
    for (int j = 0; j < 4; ++j) {
      const int r = arow + j * 32;
      v8bf v = {};
      if (r < rows) v = *(const v8bf*)(H + (size_t)(row0 + r) * INTER + k0 + ac8);
      rha[j] = v;
    }
  };
  auto storeA = [&](int b) {
#pragma unroll
    for (int j = 0; j < 4; ++j)
      *(v8bf*)&sA[b][arow + j * 32][ac8] = rha[j];
  };
#endif

  auto loadB = [&](int k0) {
#pragma unroll
    for (int i = 0; i < 4; ++i)
      rb[i] = *(const float4*)(wd + (size_t)(k0 + bkg + i) * HID + n0 + bng);
  };
  auto storeB = [&](int b) {
#pragma unroll
    for (int j = 0; j < 4; ++j) {
      v4bf v;
      v.x = (bf16_t)((const float*)&rb[0])[j];
      v.y = (bf16_t)((const float*)&rb[1])[j];
      v.z = (bf16_t)((const float*)&rb[2])[j];
      v.w = (bf16_t)((const float*)&rb[3])[j];
      *(v4bf*)&sB[b][bng + j][bkg] = v;
    }
  };

  const v8f vzero = {};
  v8f acc[2][2];
#pragma unroll
  for (int i = 0; i < 2; ++i)
#pragma unroll
    for (int j = 0; j < 2; ++j) acc[i][j] = vzero;

  auto compute = [&](int b) {
#pragma unroll
    for (int ks = 0; ks < TK; ks += 32) {
      FragAB a[2], bb[2];
#pragma unroll
      for (int mt = 0; mt < 2; ++mt) {
        const bf16_t* p = &sA[b][wm * 32 + mt * 16 + lrow][ks + half * 16];
        a[mt].h[0] = *(const v8bf*)p;
        a[mt].h[1] = *(const v8bf*)(p + 8);
      }
#pragma unroll
      for (int nt = 0; nt < 2; ++nt) {
        const bf16_t* pb = &sB[b][wn * 32 + nt * 16 + lrow][ks + half * 16];
        bb[nt].h[0] = *(const v8bf*)pb;
        bb[nt].h[1] = *(const v8bf*)(pb + 8);
      }
#pragma unroll
      for (int mt = 0; mt < 2; ++mt)
#pragma unroll
        for (int nt = 0; nt < 2; ++nt)
          acc[mt][nt] = __builtin_amdgcn_wmma_f32_16x16x32_bf16(
              false, a[mt].v, false, bb[nt].v, (short)0, acc[mt][nt], false, false);
    }
  };

  const int NK = INTER / TK;
#if HAVE_TDM
  tdmA(0, 0);
  loadB(0); storeB(0);
  wait_tensorcnt0();                       // TENSORcnt==0 before first barrier
  for (int kt = 0; kt < NK; ++kt) {
    __syncthreads();
    const int cur = kt & 1;
    if (kt + 1 < NK) {
      tdmA((kt + 1) * TK, cur ^ 1);        // async DMA into the idle buffer
      loadB((kt + 1) * TK);
      if (kt + 2 < NK)
        __builtin_prefetch(wd + (size_t)((kt + 2) * TK + bkg) * HID + n0 + bng, 0, 1);
    }
    compute(cur);                          // WMMA hides DMA + weight loads
    if (kt + 1 < NK) { storeB(cur ^ 1); wait_tensorcnt0(); }
  }
#else
  loadA(0); loadB(0);
  storeA(0); storeB(0);
  for (int kt = 0; kt < NK; ++kt) {
    __syncthreads();
    const int cur = kt & 1;
    if (kt + 1 < NK) {
      loadA((kt + 1) * TK);
      loadB((kt + 1) * TK);
      if (kt + 2 < NK)
        __builtin_prefetch(wd + (size_t)((kt + 2) * TK + bkg) * HID + n0 + bng, 0, 1);
    }
    compute(cur);
    if (kt + 1 < NK) { storeA(cur ^ 1); storeB(cur ^ 1); }
  }
#endif

  // ---- epilogue: scale by probs[row], non-temporal f32 stores -------------
#pragma unroll
  for (int mt = 0; mt < 2; ++mt)
#pragma unroll
    for (int nt = 0; nt < 2; ++nt)
#pragma unroll
      for (int r = 0; r < 8; ++r) {
        const int lm = wm * 32 + mt * 16 + half * 8 + r;
        if (lm < rows) {
          const int grow = row0 + lm;
          const int gn   = n0 + wn * 32 + nt * 16 + lrow;
          __builtin_nontemporal_store(probs[grow] * acc[mt][nt][r],
                                      Out + (size_t)grow * HID + gn);
        }
      }
}

// ---------------------------------------------------------------------------
extern "C" void kernel_launch(void* const* d_in, const int* in_sizes, int n_in,
                              void* d_out, int out_size, void* d_ws, size_t ws_size,
                              hipStream_t stream) {
  const float* X     = (const float*)d_in[0];  // [TOT, HID]
  const float* probs = (const float*)d_in[1];  // [TOT]
  const float* Wg    = (const float*)d_in[2];  // [E, HID, INTER]
  const float* Wu    = (const float*)d_in[3];  // [E, HID, INTER]
  const float* Wd    = (const float*)d_in[4];  // [E, INTER, HID]
  const int*   tpe   = (const int*)d_in[5];    // [E]
  float*       Out   = (float*)d_out;          // [TOT, HID]

  int*    offs = (int*)d_ws;                          // 9 ints
  bf16_t* H    = (bf16_t*)((char*)d_ws + 256);        // [TOT, INTER] bf16 (~23 MB)

  const int mtiles = TOT / TM + E;  // worst-case ragged tile count (72)

  moe_offsets_kernel<<<1, 32, 0, stream>>>(tpe, offs);
  moe_gemm1_kernel<<<dim3(mtiles, INTER / TN), 256, 0, stream>>>(X, Wg, Wu, offs, H);
  moe_gemm2_kernel<<<dim3(mtiles, HID / TN), 256, 0, stream>>>(H, Wd, probs, offs, Out);
}
